// LocalAttention_87832081203397
// MI455X (gfx1250) — matrix operand order repaired and checked by reading, not compiled, not run on hardware
//
#include <hip/hip_runtime.h>

// ---------------------------------------------------------------------------
// MI455X (gfx1250) local-attention:  y = softmax(mask(qk^T/sqrt(C))) v, with
// q/k/v/o projections.  All matmuls via v_wmma_f32_16x16x32_bf16 (wave32).
// Working set (~90MB) is L2-resident (192MB L2, 23.3TB/s HBM) -> compute
// bound; bf16 WMMA with fp32 accumulate.  GEMM stages LDS tiles with
// double-buffered global_load_async_to_lds_b128 (ASYNCcnt pipelining).
// ---------------------------------------------------------------------------

typedef __attribute__((ext_vector_type(16))) __bf16 v16bf;
typedef __attribute__((ext_vector_type(8)))  float  v8f;

union Frag {
    v16bf v;
    uint4 q[2];
    unsigned short s[16];
};

__device__ inline __bf16 f2bf(float f) {
    union { float f; unsigned u; } x; x.f = f;
    unsigned r = (x.u + 0x7FFFu + ((x.u >> 16) & 1u)) >> 16;   // RNE
    union { unsigned short s; __bf16 b; } y; y.s = (unsigned short)r;
    return y.b;
}

__device__ inline v8f vzero8() {
    v8f z;
#pragma unroll
    for (int i = 0; i < 8; ++i) z[i] = 0.0f;
    return z;
}

// A-fragment (16x32 bf16, A[m][k] row-major, stride ld elements):
// lane owns row (lane&15); halves h=0..7 -> K=kb+h, h=8..15 -> K=kb+16+(h-8),
// kb=(lane>>4)*8  => two contiguous 16B chunks per lane.
__device__ inline v16bf load_frag_a(const __bf16* base, int ld) {
    int lane = threadIdx.x & 31;
    const __bf16* p = base + (size_t)(lane & 15) * ld + ((lane >> 4) << 3);
    Frag f;
    f.q[0] = *(const uint4*)p;
    f.q[1] = *(const uint4*)(p + 16);
    return f.v;
}

// B-fragment (32x16 bf16, B[k][n]; "base" is row-major in n with K contiguous,
// i.e. base[n*ld + k]): lane owns col (lane&15); h=0..7 -> K=kb+h,
// h=8..15 -> K=kb+16+(h-8), kb=(lane>>4)*8 => two 16B chunks per lane.
__device__ inline v16bf load_frag_b(const __bf16* base, int ld) {
    int lane = threadIdx.x & 31;
    const __bf16* p = base + (size_t)(lane & 15) * ld + ((lane >> 4) << 3);
    Frag f;
    f.q[0] = *(const uint4*)p;
    f.q[1] = *(const uint4*)(p + 16);
    return f.v;
}

// gfx1250 async global->LDS copy (16B per lane), tracked by ASYNCcnt.
__device__ inline void async_copy_b128(unsigned lds_byte_off, const void* gsrc) {
    asm volatile("global_load_async_to_lds_b128 %0, %1, off"
                 :: "v"(lds_byte_off), "v"(gsrc) : "memory");
}
// Async loads complete in order: waiting <=N leaves the newest N in flight.
__device__ inline void wait_async0() {
    asm volatile("s_wait_asynccnt 0" ::: "memory");
}
__device__ inline void wait_async4() {
    asm volatile("s_wait_asynccnt 4" ::: "memory");
}

// ---------------------------------------------------------------------------
// fp32 -> bf16 elementwise convert
// ---------------------------------------------------------------------------
__global__ void cvt_f32_bf16(const float* __restrict__ in,
                             __bf16* __restrict__ out, int n) {
    int i = blockIdx.x * blockDim.x + threadIdx.x;
    if (i < n) out[i] = f2bf(in[i]);
}

// ---------------------------------------------------------------------------
// C[M][N] = A[M][K] @ W[N][K]^T + bias    (A,W bf16; C fp32 or bf16)
// 128x128x32 tiles, 8 waves (2x4), each wave: 4x2 tiles of 16x16.
// Double-buffered LDS staging via async-to-LDS (4 async b128 per wave per
// slab); global_prefetch one slab beyond the in-flight stage.
// ---------------------------------------------------------------------------
#define BM 128
#define BN 128
#define BK 32
#define LDT 40   // LDS row stride in elements (BK + 8 pad -> conflict free)

// Issue this thread's share (2x A + 2x W, 16B each) of one K-slab stage.
__device__ inline void stage_slab(const __bf16* __restrict__ A,
                                  const __bf16* __restrict__ W,
                                  unsigned asBase, unsigned wsBase,
                                  int m0, int n0, int k0, int K, int tid) {
#pragma unroll
    for (int s = 0; s < 2; ++s) {                 // 512 16B units / 256 threads
        int u   = tid + (s << 8);
        int row = u >> 2;
        int qo  = (u & 3) << 3;                   // element offset 0/8/16/24
        unsigned loff = (unsigned)(row * LDT + qo) * 2u;
        const __bf16* ag = A + (size_t)(m0 + row) * K + k0 + qo;
        const __bf16* wg = W + (size_t)(n0 + row) * K + k0 + qo;
        async_copy_b128(asBase + loff, ag);
        async_copy_b128(wsBase + loff, wg);
        __builtin_prefetch(ag + BK, 0, 3);        // one slab beyond the stage
        __builtin_prefetch(wg + BK, 0, 3);
    }
}

__global__ __launch_bounds__(256)
void gemm_xWT(const __bf16* __restrict__ A, const __bf16* __restrict__ W,
              const float* __restrict__ bias, float* __restrict__ Cf,
              __bf16* __restrict__ Cb, int M, int N, int K) {
    __shared__ __bf16 As[2][BM * LDT];
    __shared__ __bf16 Ws[2][BN * LDT];

    const int tid = threadIdx.x;
    const int wave = tid >> 5, lane = tid & 31;
    const int wm = wave >> 2, wn = wave & 3;          // 2x4 wave grid
    const int m0 = blockIdx.y * BM, n0 = blockIdx.x * BN;

    v8f acc[4][2];
#pragma unroll
    for (int mt = 0; mt < 4; ++mt)
#pragma unroll
        for (int nt = 0; nt < 2; ++nt) acc[mt][nt] = vzero8();

    const unsigned asBase[2] = { (unsigned)(size_t)(&As[0][0]),
                                 (unsigned)(size_t)(&As[1][0]) };
    const unsigned wsBase[2] = { (unsigned)(size_t)(&Ws[0][0]),
                                 (unsigned)(size_t)(&Ws[1][0]) };

    // prologue: stage slab 0 into buffer 0
    stage_slab(A, W, asBase[0], wsBase[0], m0, n0, 0, K, tid);

    for (int k0 = 0; k0 < K; k0 += BK) {
        const int cur = (k0 >> 5) & 1;
        if (k0 + BK < K) {
            // stage next slab into the other buffer, then wait only for the
            // current slab's 4 copies (in-order completion) -> DMA overlaps
            // with the WMMA stream below.
            stage_slab(A, W, asBase[cur ^ 1], wsBase[cur ^ 1],
                       m0, n0, k0 + BK, K, tid);
            wait_async4();
        } else {
            wait_async0();
        }
        __syncthreads();                         // all waves' copies landed

        v16bf af[4], bf[2];
#pragma unroll
        for (int mt = 0; mt < 4; ++mt)
            af[mt] = load_frag_a(&As[cur][(wm * 64 + mt * 16) * LDT], LDT);
#pragma unroll
        for (int nt = 0; nt < 2; ++nt)
            bf[nt] = load_frag_b(&Ws[cur][(wn * 32 + nt * 16) * LDT], LDT);
#pragma unroll
        for (int mt = 0; mt < 4; ++mt)
#pragma unroll
            for (int nt = 0; nt < 2; ++nt)
                acc[mt][nt] = __builtin_amdgcn_wmma_f32_16x16x32_bf16(
                    false, af[mt], false, bf[nt], (short)0, acc[mt][nt],
                    false, false);
        __syncthreads();                         // reads done before refill
    }

    // epilogue: D layout -> row M = (lane>>4)*8 + r, col N = lane&15
#pragma unroll
    for (int mt = 0; mt < 4; ++mt) {
        int mbase = m0 + wm * 64 + mt * 16 + ((lane >> 4) << 3);
#pragma unroll
        for (int nt = 0; nt < 2; ++nt) {
            int col = n0 + wn * 32 + nt * 16 + (lane & 15);
            float bv = bias[col];
#pragma unroll
            for (int r = 0; r < 8; ++r) {
                float v = acc[mt][nt][r] + bv;
                size_t idx = (size_t)(mbase + r) * N + col;
                if (Cf) Cf[idx] = v;
                else    Cb[idx] = f2bf(v);
            }
        }
    }
}

// ---------------------------------------------------------------------------
// Local attention, flash-style.  One block = one (batch, 16-row query tile).
// 8 waves split C=1024 into 128-wide slices; keys streamed in 32-wide chunks.
// Mask keeps j >= i-16 (anti-causal + 16 band), so chunks start at i0-16
// rounded down to 32; only the first chunk is partially masked.
// ---------------------------------------------------------------------------
#define TT 2048
#define CCH 1024
#define WIN 16

__global__ __launch_bounds__(256)
void attn_local(const __bf16* __restrict__ Q, const __bf16* __restrict__ Km,
                const __bf16* __restrict__ V, __bf16* __restrict__ Y) {
    __shared__ float  Ssh[16 * 33];   // cross-wave S reduction (16x32, pad)
    __shared__ __bf16 Psh[16 * 40];   // exp(S-m) tile, A-frag layout friendly
    __shared__ float  abuf[16];       // per-row rescale alpha
    __shared__ float  lbuf[16];       // per-row running denom

    const int tid  = threadIdx.x;
    const int wave = tid >> 5, lane = tid & 31;
    const int i0   = blockIdx.x * 16;
    const int b    = blockIdx.y;
    const int cs   = wave * 128;                 // this wave's C slice
    const int mrow = (lane >> 4) << 3;           // D-tile row base for this lane
    const int ncol = lane & 15;                  // D-tile col for this lane
    const float scale = 0.03125f;                // 1/sqrt(1024)

    // q fragments for this wave's C slice (fixed across key chunks)
    v16bf qf[4];
#pragma unroll
    for (int p = 0; p < 4; ++p)
        qf[p] = load_frag_a(Q + (size_t)(b * TT + i0) * CCH + cs + p * 32, CCH);

    v8f o[8];
#pragma unroll
    for (int t = 0; t < 8; ++t) o[t] = vzero8();

    float m_r = -1e30f, l_r = 0.0f;              // wave0 lanes 0..15 own row stats

    const int jlow   = i0 - WIN;
    const int jstart = jlow > 0 ? (jlow & ~31) : 0;

    for (int j0 = jstart; j0 < TT; j0 += 32) {
        // pull next chunk's K/V rows toward this WGP while we work
        if (j0 + 32 < TT) {
            int pr = j0 + 32 + (lane & 31);      // 32 rows, one per lane
            __builtin_prefetch(Km + (size_t)(b * TT + pr) * CCH + cs, 0, 3);
            __builtin_prefetch(V  + (size_t)(b * TT + pr) * CCH + cs, 0, 3);
        }

        // zero the shared S accumulator
        for (int i = tid; i < 16 * 33; i += 256) Ssh[i] = 0.0f;
        __syncthreads();

        // partial S = q k^T over this wave's C slice (8 WMMAs)
        v8f sacc[2];
#pragma unroll
        for (int jt = 0; jt < 2; ++jt) sacc[jt] = vzero8();
#pragma unroll
        for (int p = 0; p < 4; ++p) {
#pragma unroll
            for (int jt = 0; jt < 2; ++jt) {
                v16bf kf = load_frag_b(
                    Km + (size_t)(b * TT + j0 + jt * 16) * CCH + cs + p * 32, CCH);
                sacc[jt] = __builtin_amdgcn_wmma_f32_16x16x32_bf16(
                    false, qf[p], false, kf, (short)0, sacc[jt], false, false);
            }
        }
#pragma unroll
        for (int jt = 0; jt < 2; ++jt)
#pragma unroll
            for (int r = 0; r < 8; ++r)
                atomicAdd(&Ssh[(mrow + r) * 33 + jt * 16 + ncol], sacc[jt][r]);
        __syncthreads();

        // wave 0: online softmax stats + P tile (bf16)
        if (wave == 0 && lane < 16) {
            const int r = lane;
            float sv[32];
            float mc = -1e30f;
#pragma unroll
            for (int jj = 0; jj < 32; ++jj) {
                float s = Ssh[r * 33 + jj] * scale;
                if (j0 + jj < i0 + r - WIN) s = -1e30f;   // mask: keep j >= i-16
                sv[jj] = s;
                mc = fmaxf(mc, s);
            }
            float mn    = fmaxf(m_r, mc);
            float alpha = __expf(m_r - mn);
            float rs    = 0.0f;
#pragma unroll
            for (int jj = 0; jj < 32; ++jj) {
                float pv = __expf(sv[jj] - mn);
                rs += pv;
                Psh[r * 40 + jj] = f2bf(pv);
            }
            l_r = l_r * alpha + rs;
            m_r = mn;
            abuf[r] = alpha;
            lbuf[r] = l_r;
        }
        __syncthreads();

        // rescale running O, then O += P @ V  (8 WMMAs over this C slice)
        float am[8];
#pragma unroll
        for (int r = 0; r < 8; ++r) am[r] = abuf[mrow + r];
#pragma unroll
        for (int t = 0; t < 8; ++t)
#pragma unroll
            for (int r = 0; r < 8; ++r) o[t][r] *= am[r];

        v16bf pf = load_frag_a(Psh, 40);
        const unsigned short* Vu = (const unsigned short*)V;
        const int kb = (lane >> 4) << 3;
#pragma unroll
        for (int nt = 0; nt < 8; ++nt) {
            int c = cs + nt * 16 + ncol;
            Frag f;
#pragma unroll
            for (int h = 0; h < 16; ++h)   // B[kk][c]: gather column of V (L2 hit)
                f.s[h] = Vu[(size_t)(b * TT + j0 + kb + (h < 8 ? h : h + 8)) * CCH + c];
            o[nt] = __builtin_amdgcn_wmma_f32_16x16x32_bf16(
                false, pf, false, f.v, (short)0, o[nt], false, false);
        }
    }

    __syncthreads();
    float rl[8];
#pragma unroll
    for (int r = 0; r < 8; ++r) rl[r] = 1.0f / lbuf[mrow + r];
#pragma unroll
    for (int nt = 0; nt < 8; ++nt)
#pragma unroll
        for (int r = 0; r < 8; ++r) {
            size_t idx = (size_t)(b * TT + i0 + mrow + r) * CCH + cs + nt * 16 + ncol;
            Y[idx] = f2bf(o[nt][r] * rl[r]);
        }
}

// ---------------------------------------------------------------------------
extern "C" void kernel_launch(void* const* d_in, const int* in_sizes, int n_in,
                              void* d_out, int out_size, void* d_ws, size_t ws_size,
                              hipStream_t stream) {
    (void)in_sizes; (void)n_in; (void)out_size; (void)ws_size;
    const float* x  = (const float*)d_in[0];
    const float* Wq = (const float*)d_in[1];
    const float* bq = (const float*)d_in[2];
    const float* Wk = (const float*)d_in[3];
    const float* bk = (const float*)d_in[4];
    const float* Wv = (const float*)d_in[5];
    const float* bv = (const float*)d_in[6];
    const float* Wo = (const float*)d_in[7];
    const float* bo = (const float*)d_in[8];
    float* out = (float*)d_out;

    const int B = 4, T = 2048, C = 1024;
    const int M = B * T;

    char* p = (char*)d_ws;
    __bf16* xb  = (__bf16*)p; p += (size_t)M * C * 2;
    __bf16* wqb = (__bf16*)p; p += (size_t)C * C * 2;
    __bf16* wkb = (__bf16*)p; p += (size_t)C * C * 2;
    __bf16* wvb = (__bf16*)p; p += (size_t)C * C * 2;
    __bf16* wob = (__bf16*)p; p += (size_t)C * C * 2;
    __bf16* qb  = (__bf16*)p; p += (size_t)M * C * 2;
    __bf16* kb  = (__bf16*)p; p += (size_t)M * C * 2;
    __bf16* vb  = (__bf16*)p; p += (size_t)M * C * 2;
    __bf16* yb  = (__bf16*)p; p += (size_t)M * C * 2;

    const int nx = M * C, nw = C * C;
    cvt_f32_bf16<<<(nx + 255) / 256, 256, 0, stream>>>(x,  xb,  nx);
    cvt_f32_bf16<<<(nw + 255) / 256, 256, 0, stream>>>(Wq, wqb, nw);
    cvt_f32_bf16<<<(nw + 255) / 256, 256, 0, stream>>>(Wk, wkb, nw);
    cvt_f32_bf16<<<(nw + 255) / 256, 256, 0, stream>>>(Wv, wvb, nw);
    cvt_f32_bf16<<<(nw + 255) / 256, 256, 0, stream>>>(Wo, wob, nw);

    dim3 gg(C / BN, M / BM);
    gemm_xWT<<<gg, 256, 0, stream>>>(xb, wqb, bq, nullptr, qb, M, C, C);
    gemm_xWT<<<gg, 256, 0, stream>>>(xb, wkb, bk, nullptr, kb, M, C, C);
    gemm_xWT<<<gg, 256, 0, stream>>>(xb, wvb, bv, nullptr, vb, M, C, C);

    dim3 ga(T / 16, B);
    attn_local<<<ga, 256, 0, stream>>>(qb, kb, vb, yb);

    gemm_xWT<<<gg, 256, 0, stream>>>(yb, wob, bo, out, nullptr, M, C, C);
}
